// PostHocRBFQuantizer_90237262889780
// MI455X (gfx1250) — compile-verified, matrix operand order checked by reading, not run on hardware
//
#include <hip/hip_runtime.h>
#include <hip/hip_bf16.h>

#define NN 32768
#define KK 4096
#define DD 512
#define GAMMA 0.005f
#define DECAY 0.99f
#define EPSV 1e-5f

#define AS1 __attribute__((address_space(1)))
#define AS3 __attribute__((address_space(3)))

#if defined(__has_builtin)
# if __has_builtin(__builtin_amdgcn_global_load_async_to_lds_b128)
#  define HAVE_ASYNC_LDS 1
# endif
# if __has_builtin(__builtin_amdgcn_s_wait_asynccnt)
#  define HAVE_WAIT_ASYNC 1
# endif
#endif

typedef __attribute__((ext_vector_type(16))) __bf16 v16bf;
typedef __attribute__((ext_vector_type(8)))  __bf16 v8bf;
typedef __attribute__((ext_vector_type(8)))  float  v8f;
typedef __attribute__((ext_vector_type(4)))  float  v4f;
typedef __attribute__((ext_vector_type(4)))  int    v4i;

union V16U { v16bf v; v8bf h[2]; };

#define TSTRIDE 536   // padded row stride (1072 B): conflict-free ds_load_b128

// ---------------------------------------------------------------------------
// Kernel 1: convert emb fp32 -> bf16 (workspace) and compute 0.5*||e_k||^2
// ---------------------------------------------------------------------------
__global__ void prep_emb_kernel(const float* __restrict__ emb,
                                __bf16* __restrict__ ebf,
                                float* __restrict__ hnorm) {
    int k = blockIdx.x;            // one row per block, 128 threads
    int t = threadIdx.x;
    const float* row = emb + (size_t)k * DD;
    __bf16* orow = ebf + (size_t)k * DD;
    float s = 0.f;
#pragma unroll
    for (int i = 0; i < 4; ++i) {
        float v = row[t + i * 128];
        orow[t + i * 128] = (__bf16)v;
        s += v * v;
    }
    __shared__ float red[128];
    red[t] = s;
    __syncthreads();
#pragma unroll
    for (int off = 64; off > 0; off >>= 1) {
        if (t < off) red[t] += red[t + off];
        __syncthreads();
    }
    if (t == 0) hnorm[k] = 0.5f * red[0];
}

// ---------------------------------------------------------------------------
// Kernel 2: init outputs: new_cs = DECAY*cs, new_ema = DECAY*ema, n_total = 0
// ---------------------------------------------------------------------------
__global__ void init_out_kernel(const float* __restrict__ cs,
                                const float* __restrict__ emaw,
                                float* __restrict__ out_cs,
                                float* __restrict__ out_ema,
                                float* __restrict__ n_total) {
    size_t i = (size_t)blockIdx.x * blockDim.x + threadIdx.x;   // over K*D
    out_ema[i] = DECAY * emaw[i];
    if (i < KK) out_cs[i] = DECAY * cs[i];
    if (i == 0) *n_total = 0.f;
}

// ---------------------------------------------------------------------------
// Async helpers
// ---------------------------------------------------------------------------
__device__ __forceinline__ void wait_async(int n) {
#if defined(HAVE_WAIT_ASYNC)
    if (n == 0) __builtin_amdgcn_s_wait_asynccnt(0);
    else        __builtin_amdgcn_s_wait_asynccnt(4);
#else
    if (n == 0) asm volatile("s_wait_asynccnt 0x0" ::: "memory");
    else        asm volatile("s_wait_asynccnt 0x4" ::: "memory");
#endif
}

#if defined(HAVE_ASYNC_LDS)
__device__ __forceinline__ void issue_tile_async(const __bf16* __restrict__ src,
                                                 __bf16* buf, int tid) {
#pragma unroll
    for (int i = 0; i < 4; ++i) {
        int chunk = i * 256 + tid;                 // 1024 x 16B = 16 KB tile
        int r   = chunk >> 6;
        int col = (chunk & 63) * 8;
        __builtin_amdgcn_global_load_async_to_lds_b128(
            (AS1 v4i*)(void*)(src + (size_t)chunk * 8),
            (AS3 v4i*)(void*)(buf + r * TSTRIDE + col),
            0, 0);
    }
}
#endif

// ---------------------------------------------------------------------------
// Kernel 3: fused GEMM + argmax via WMMA bf16.
// Block = 256 threads (8 waves). Wave w owns 16 rows of z (A resident in
// VGPRs as 16 x v16bf fragments). All waves share a double-buffered 16x512
// bf16 emb tile in LDS, filled by GLOBAL_LOAD_ASYNC_TO_LDS_B128 so tile kt+1
// streams in while tile kt runs through 16 chained WMMAs.
// score = z.e - 0.5||e||^2 ; running argmax per row.
// ---------------------------------------------------------------------------
__global__ void __launch_bounds__(256, 1)
argmax_wmma_kernel(const float* __restrict__ z,
                   const __bf16* __restrict__ ebf,
                   const float* __restrict__ hnorm,
                   int* __restrict__ out_idx) {
    __shared__ __bf16 etile[2][16 * TSTRIDE];
    __shared__ float  s_hn[KK];          // 16 KB half-norm table

    const int tid  = threadIdx.x;
    const int lane = tid & 31;
    const int wave = tid >> 5;
    const int mloc = lane & 15;
    const int kb   = (lane >> 4) * 8;    // K-subgroup select per WMMA layout
    const int base_m = blockIdx.x * 128 + wave * 16;
    const int NT = KK / 16;

    for (int i = tid; i < KK; i += 256) s_hn[i] = hnorm[i];

    // ---- load A fragments: 16 rows x 512 cols, bf16, register resident ----
    const float* zrow = z + (size_t)(base_m + mloc) * DD;
    v16bf afrag[16];
#pragma unroll
    for (int c = 0; c < 16; ++c) {
        const int d0 = c * 32 + kb;
        v4f f0 = *(const v4f*)(zrow + d0);
        v4f f1 = *(const v4f*)(zrow + d0 + 4);
        v4f f2 = *(const v4f*)(zrow + d0 + 16);
        v4f f3 = *(const v4f*)(zrow + d0 + 20);
        v16bf a;
#pragma unroll
        for (int j = 0; j < 4; ++j) {
            a[j]      = (__bf16)f0[j];
            a[4 + j]  = (__bf16)f1[j];
            a[8 + j]  = (__bf16)f2[j];
            a[12 + j] = (__bf16)f3[j];
        }
        afrag[c] = a;
    }

    float bv[8];
    int   bidx[8];
#pragma unroll
    for (int r = 0; r < 8; ++r) { bv[r] = -3.402823466e38f; bidx[r] = 0; }

#if defined(HAVE_ASYNC_LDS)
    issue_tile_async(ebf, &etile[0][0], tid);          // preload tile 0
#endif

    for (int kt = 0; kt < NT; ++kt) {
        const int cur = kt & 1;
#if defined(HAVE_ASYNC_LDS)
        const bool more = (kt + 1 < NT);
        if (more)
            issue_tile_async(ebf + (size_t)(kt + 1) * 16 * DD,
                             &etile[cur ^ 1][0], tid); // overlap with compute
        wait_async(more ? 4 : 0);                      // tile kt has landed
        __syncthreads();                               // visible to all waves
#else
        {   // fallback: stage through VGPRs
            const v8bf* src = (const v8bf*)(ebf + (size_t)kt * 16 * DD);
#pragma unroll
            for (int i = 0; i < 4; ++i) {
                int chunk = i * 256 + tid;
                int r   = chunk >> 6;
                int col = (chunk & 63) * 8;
                *(v8bf*)(&etile[0][r * TSTRIDE + col]) = src[chunk];
            }
            __syncthreads();
            if (kt + 1 < NT)
                __builtin_prefetch((const char*)(ebf + (size_t)(kt + 1) * 16 * DD) + tid * 16, 0, 1);
        }
#endif

#if defined(HAVE_ASYNC_LDS)
        const __bf16* brow = &etile[cur][mloc * TSTRIDE];
#else
        const __bf16* brow = &etile[0][mloc * TSTRIDE];
#endif
        v8f acc = {};
#pragma unroll
        for (int c = 0; c < 16; ++c) {
            const int d0 = c * 32 + kb;
            V16U b;
            b.h[0] = *(const v8bf*)(brow + d0);
            b.h[1] = *(const v8bf*)(brow + d0 + 16);
            acc = __builtin_amdgcn_wmma_f32_16x16x32_bf16(
                false, afrag[c], false, b.v, (short)0, acc, false, false);
        }

        const int n_code = kt * 16 + mloc;
        const float hn = s_hn[n_code];
#pragma unroll
        for (int r = 0; r < 8; ++r) {
            float val = acc[r] - hn;
            if (val > bv[r]) { bv[r] = val; bidx[r] = n_code; }
        }
        __syncthreads();   // all waves done reading this buffer
    }

    // cross-lane argmax within each 16-lane half (xor 1,2,4,8 keeps halves)
#pragma unroll
    for (int r = 0; r < 8; ++r) {
        float v = bv[r];
        int   bi = bidx[r];
#pragma unroll
        for (int off = 1; off < 16; off <<= 1) {
            float ov = __shfl_xor(v, off, 32);
            int   oi = __shfl_xor(bi, off, 32);
            if (ov > v || (ov == v && oi < bi)) { v = ov; bi = oi; }
        }
        if (mloc == 0)    // lanes 0 and 16 cover M = r and M = r + 8
            out_idx[base_m + r + 8 * (lane >> 4)] = bi;
    }
}

// ---------------------------------------------------------------------------
// Kernel 4: scatter: new_ema[idx[n]] += (1-DECAY)*z[n]; new_cs[idx[n]] += ...
// ---------------------------------------------------------------------------
__global__ void scatter_kernel(const float* __restrict__ z,
                               const int* __restrict__ idx,
                               float* __restrict__ out_ema,
                               float* __restrict__ out_cs) {
    int t = threadIdx.x;
    int row = blockIdx.x * 4 + (t >> 6);
    int l = t & 63;
    int k = idx[row];
    const float* zr = z + (size_t)row * DD;
    float* er = out_ema + (size_t)k * DD;
    const float a = 1.0f - DECAY;
#pragma unroll
    for (int j = 0; j < 8; ++j) {
        int d = l + j * 64;
        atomicAdd(&er[d], a * zr[d]);
    }
    if (l == 0) atomicAdd(&out_cs[k], a);
}

// ---------------------------------------------------------------------------
// Kernel 5: n_total = sum(new_cluster_size)
// ---------------------------------------------------------------------------
__global__ void reduce_n_kernel(const float* __restrict__ out_cs,
                                float* __restrict__ n_total) {
    __shared__ float red[256];
    int t = threadIdx.x;
    red[t] = out_cs[blockIdx.x * 256 + t];
    __syncthreads();
#pragma unroll
    for (int off = 128; off > 0; off >>= 1) {
        if (t < off) red[t] += red[t + off];
        __syncthreads();
    }
    if (t == 0) atomicAdd(n_total, red[0]);
}

// ---------------------------------------------------------------------------
// Kernel 6: new_weight = new_ema / ((new_cs+eps)/(n+K+eps)*n)
// ---------------------------------------------------------------------------
__global__ void finalize_kernel(const float* __restrict__ out_ema,
                                const float* __restrict__ out_cs,
                                const float* __restrict__ n_total,
                                float* __restrict__ out_w) {
    size_t i = (size_t)blockIdx.x * blockDim.x + threadIdx.x;
    int k = (int)(i >> 9);
    float n = *n_total;
    float stab = (out_cs[k] + EPSV) / (n + (float)KK + EPSV) * n;
    out_w[i] = out_ema[i] / stab;
}

// ---------------------------------------------------------------------------
extern "C" void kernel_launch(void* const* d_in, const int* in_sizes, int n_in,
                              void* d_out, int out_size, void* d_ws, size_t ws_size,
                              hipStream_t stream) {
    const float* z       = (const float*)d_in[0];
    const float* emb_w   = (const float*)d_in[1];
    const float* cs      = (const float*)d_in[2];
    const float* ema_w   = (const float*)d_in[3];

    float* out     = (float*)d_out;
    int*   out_idx = (int*)out;                       // [N] int32
    float* out_w   = out + NN;                        // [K*D]
    float* out_cs  = out + NN + (size_t)KK * DD;      // [K]
    float* out_ema = out_cs + KK;                     // [K*D]

    char* ws = (char*)d_ws;
    __bf16* ebf    = (__bf16*)ws;                                    // 4 MB
    float*  hnorm  = (float*)(ws + (size_t)KK * DD * 2);             // 16 KB
    float*  ntotal = (float*)(ws + (size_t)KK * DD * 2 + KK * 4);    // 4 B

    prep_emb_kernel<<<KK, 128, 0, stream>>>(emb_w, ebf, hnorm);
    init_out_kernel<<<(KK * DD) / 256, 256, 0, stream>>>(cs, ema_w, out_cs, out_ema, ntotal);
    argmax_wmma_kernel<<<NN / 128, 256, 0, stream>>>(z, ebf, hnorm, out_idx);
    scatter_kernel<<<NN / 4, 256, 0, stream>>>(z, out_idx, out_ema, out_cs);
    reduce_n_kernel<<<KK / 256, 256, 0, stream>>>(out_cs, ntotal);
    finalize_kernel<<<(KK * DD) / 256, 256, 0, stream>>>(out_ema, out_cs, ntotal, out_w);
}
